// LocalSelfAttentionFusion_12893491823061
// MI455X (gfx1250) — matrix operand order repaired and checked by reading, not compiled, not run on hardware
//
#include <hip/hip_runtime.h>

// ---------------------------------------------------------------------------
// Fused 2-token local self-attention for MI455X (gfx1250, wave32, WMMA)
//   x(M,2,384) -> qkv = x@Win^T+b -> 8-head 2x2 attention -> y@Wout^T+b
//   -> LayerNorm -> mean-pool tokens -> out(M,384)
// GEMMs run on v_wmma_f32_16x16x32_bf16 with fp32 accumulation.
// ---------------------------------------------------------------------------

typedef __attribute__((ext_vector_type(16))) __bf16 v16bf;
typedef __attribute__((ext_vector_type(8)))  __bf16 v8bf;
typedef __attribute__((ext_vector_type(8)))  float  v8f;
typedef __attribute__((ext_vector_type(4)))  float  v4f;

#define NHEAD   8
#define DIM     384
#define HD      48
#define TD      1152            // 3*DIM
#define ROWS    64              // rows per workgroup  (= 2 * VOXPB)
#define VOXPB   32              // voxels per workgroup
#define XS      400             // X/Y LDS row stride (bf16 elems, padded)
#define QS      1168            // qkv LDS row stride (bf16 elems, padded)
#define OS      388             // out LDS row stride (f32 elems, padded)
#define KC      (DIM / 32)      // 12 K-chunks of 32
#define LDS_X_BYTES   (ROWS * XS * 2)        // 51200
#define LDS_QKV_BYTES (ROWS * QS * 2)        // 149504
#define LDS_TOTAL     (LDS_X_BYTES + LDS_QKV_BYTES)  // 200704 (< 320KB/WG)
#define WIN_ELEMS  (TD * DIM)   // 442368
#define WOUT_ELEMS (DIM * DIM)  // 147456

__device__ __forceinline__ unsigned short f2bf(float f) {
  unsigned u = __float_as_uint(f);
  u += 0x7FFFu + ((u >> 16) & 1u);            // round-to-nearest-even
  return (unsigned short)(u >> 16);
}
__device__ __forceinline__ float bf2f(unsigned short b) {
  return __uint_as_float(((unsigned)b) << 16);
}

union V16U { v16bf v; v8bf h[2]; };

// A-frag (16x32 bf16, M x K) from LDS: lanes 0-15 hold row
// m=lane with K {0..15}; lanes 16-31 hold K {16..31}.
__device__ __forceinline__ v16bf load_a(const unsigned short* lds, int rowBase,
                                        int kb, int lane) {
  int row = rowBase + (lane & 15);
  int hi  = lane >> 4;
  const unsigned short* p = lds + row * XS + kb + hi * 16;
  V16U u;
  u.h[0] = *(const v8bf*)(p);
  u.h[1] = *(const v8bf*)(p + 8);
  return u.v;
}

// B-frag (32x16 bf16, K x N) from global bf16 W (row-major [N][K-dim]):
// B[k,n] = W[n,k] -> lane n%16 reads 16 contiguous elems of W row n,
// lane-half selects K 0..15 vs 16..31.
__device__ __forceinline__ v16bf load_b(const unsigned short* w, int ncol0,
                                        int kb, int lane) {
  int col = ncol0 + (lane & 15);
  int ko  = kb + ((lane >> 4) << 4);
  return *(const v16bf*)(w + (size_t)col * DIM + ko);
}

__device__ __forceinline__ v8f wmma_bf16(v16bf a, v16bf b, v8f c) {
  return __builtin_amdgcn_wmma_f32_16x16x32_bf16(
      /*neg_a=*/false, a, /*neg_b=*/false, b,
      /*c_mod=*/(short)0, c, /*reuse_a=*/false, /*reuse_b=*/false);
}

__global__ void wcvt_kernel(const float* __restrict__ win,
                            const float* __restrict__ wout,
                            unsigned short* __restrict__ ws) {
  const int total = WIN_ELEMS + WOUT_ELEMS;
  for (int i = blockIdx.x * blockDim.x + threadIdx.x; i < total;
       i += gridDim.x * blockDim.x) {
    float f = (i < WIN_ELEMS) ? win[i] : wout[i - WIN_ELEMS];
    ws[i] = f2bf(f);
  }
}

__global__ __launch_bounds__(256, 1)
void attn_fused_kernel(const float* __restrict__ feat1,
                       const float* __restrict__ feat2,
                       const float* __restrict__ b_in,
                       const float* __restrict__ b_out,
                       const float* __restrict__ ln_w,
                       const float* __restrict__ ln_b,
                       const unsigned short* __restrict__ win_bf,
                       const unsigned short* __restrict__ wout_bf,
                       float* __restrict__ out) {
  extern __shared__ unsigned char smem[];
  unsigned short* Xl  = (unsigned short*)smem;                   // ROWS x XS (also Y)
  unsigned short* QKV = (unsigned short*)(smem + LDS_X_BYTES);   // ROWS x QS
  float*          Ol  = (float*)(smem + LDS_X_BYTES);            // ROWS x OS overlay

  const int tid  = threadIdx.x;
  const int lane = tid & 31;
  const int wv   = tid >> 5;                 // wave 0..7
  const int vox0 = blockIdx.x * VOXPB;

  // ---- Stage A: interleave feat1/feat2 rows into X (bf16, LDS) -------------
  for (int i = tid; i < (ROWS * DIM) / 4; i += 256) {
    int e   = i * 4;
    int row = e / DIM;
    int col = e - row * DIM;
    const float* src =
        ((row & 1) ? feat2 : feat1) + (size_t)(vox0 + (row >> 1)) * DIM + col;
    v4f f = *(const v4f*)src;
    unsigned short* dst = Xl + row * XS + col;
    dst[0] = f2bf(f.x); dst[1] = f2bf(f.y);
    dst[2] = f2bf(f.z); dst[3] = f2bf(f.w);
  }
  __syncthreads();

  // ---- Stage B: GEMM1  qkv = X @ Win^T + b_in  (72 ntiles, 8 waves x 9) ----
  for (int blk = 0; blk < 3; ++blk) {
    const int nt0 = wv * 9 + blk * 3;
    v8f acc[3][4];
#pragma unroll
    for (int n = 0; n < 3; ++n)
#pragma unroll
      for (int m = 0; m < 4; ++m)
        acc[n][m] = (v8f){0.f, 0.f, 0.f, 0.f, 0.f, 0.f, 0.f, 0.f};

    for (int kc = 0; kc < KC; ++kc) {
      const int kb = kc * 32;
      if (kc + 1 < KC)   // warm L2->L0 path for next K-chunk of W
        __builtin_prefetch(win_bf + (size_t)(nt0 * 16 + (lane & 15)) * DIM + kb + 32, 0, 0);
      v16bf bfr[3];
#pragma unroll
      for (int n = 0; n < 3; ++n) bfr[n] = load_b(win_bf, (nt0 + n) * 16, kb, lane);
#pragma unroll
      for (int m = 0; m < 4; ++m) {
        v16bf af = load_a(Xl, m * 16, kb, lane);
#pragma unroll
        for (int n = 0; n < 3; ++n) acc[n][m] = wmma_bf16(af, bfr[n], acc[n][m]);
      }
    }
    // epilogue: +bias, cvt bf16, scatter into QKV
#pragma unroll
    for (int n = 0; n < 3; ++n) {
      const int col  = (nt0 + n) * 16 + (lane & 15);
      const float bi = b_in[col];
#pragma unroll
      for (int m = 0; m < 4; ++m)
#pragma unroll
        for (int j = 0; j < 8; ++j) {
          int row = m * 16 + j + ((lane >> 4) << 3);
          QKV[row * QS + col] = f2bf(acc[n][m][j] + bi);
        }
    }
  }
  __syncthreads();

  // ---- Stage C: per-(voxel,head) 2x2 attention; Y overwrites X buffer ------
  {
    const int h  = tid & 7;
    const int vx = tid >> 3;                 // 0..31
    const int r0 = 2 * vx, r1 = r0 + 1;
    const unsigned short* q0 = QKV + r0 * QS + h * HD;
    const unsigned short* q1 = QKV + r1 * QS + h * HD;
    const unsigned short* k0 = QKV + r0 * QS + DIM + h * HD;
    const unsigned short* k1 = QKV + r1 * QS + DIM + h * HD;
    const unsigned short* v0 = QKV + r0 * QS + 2 * DIM + h * HD;
    const unsigned short* v1 = QKV + r1 * QS + 2 * DIM + h * HD;
    float s00 = 0.f, s01 = 0.f, s10 = 0.f, s11 = 0.f;
#pragma unroll 4
    for (int d = 0; d < HD; ++d) {
      float a0 = bf2f(q0[d]), a1 = bf2f(q1[d]);
      float c0 = bf2f(k0[d]), c1 = bf2f(k1[d]);
      s00 += a0 * c0; s01 += a0 * c1; s10 += a1 * c0; s11 += a1 * c1;
    }
    const float scale = 0.14433756729740643f;  // 1/sqrt(48)
    s00 *= scale; s01 *= scale; s10 *= scale; s11 *= scale;
    float m0 = fmaxf(s00, s01), m1 = fmaxf(s10, s11);
    float e00 = __expf(s00 - m0), e01 = __expf(s01 - m0);
    float e10 = __expf(s10 - m1), e11 = __expf(s11 - m1);
    float a00 = e00 / (e00 + e01), a01 = e01 / (e00 + e01);
    float a10 = e10 / (e10 + e11), a11 = e11 / (e10 + e11);
    unsigned short* y0 = Xl + r0 * XS + h * HD;
    unsigned short* y1 = Xl + r1 * XS + h * HD;
#pragma unroll 4
    for (int d = 0; d < HD; ++d) {
      float vv0 = bf2f(v0[d]), vv1 = bf2f(v1[d]);
      y0[d] = f2bf(a00 * vv0 + a01 * vv1);
      y1[d] = f2bf(a10 * vv0 + a11 * vv1);
    }
  }
  __syncthreads();

  // ---- Stage D: GEMM2  o = Y @ Wout^T + b_out -> fp32 LDS (overlays QKV) ---
  {
    const int nt0 = wv * 3;                  // 24 ntiles, 3 per wave
    v8f acc[3][4];
#pragma unroll
    for (int n = 0; n < 3; ++n)
#pragma unroll
      for (int m = 0; m < 4; ++m)
        acc[n][m] = (v8f){0.f, 0.f, 0.f, 0.f, 0.f, 0.f, 0.f, 0.f};

    for (int kc = 0; kc < KC; ++kc) {
      const int kb = kc * 32;
      v16bf bfr[3];
#pragma unroll
      for (int n = 0; n < 3; ++n) bfr[n] = load_b(wout_bf, (nt0 + n) * 16, kb, lane);
#pragma unroll
      for (int m = 0; m < 4; ++m) {
        v16bf af = load_a(Xl, m * 16, kb, lane);
#pragma unroll
        for (int n = 0; n < 3; ++n) acc[n][m] = wmma_bf16(af, bfr[n], acc[n][m]);
      }
    }
#pragma unroll
    for (int n = 0; n < 3; ++n) {
      const int col  = (nt0 + n) * 16 + (lane & 15);
      const float bo = b_out[col];
#pragma unroll
      for (int m = 0; m < 4; ++m)
#pragma unroll
        for (int j = 0; j < 8; ++j) {
          int row = m * 16 + j + ((lane >> 4) << 3);
          Ol[row * OS + col] = acc[n][m][j] + bo;
        }
    }
  }
  __syncthreads();

  // ---- Stage E: LayerNorm per row + mean-pool token pair -> global ---------
  for (int vi = 0; vi < 4; ++vi) {
    const int vx = wv * 4 + vi;
    const int r0 = 2 * vx, r1 = r0 + 1;
    float s0 = 0.f, q0 = 0.f, s1 = 0.f, q1 = 0.f;
#pragma unroll
    for (int i = 0; i < 12; ++i) {
      int d = lane + 32 * i;
      float y0 = Ol[r0 * OS + d], y1 = Ol[r1 * OS + d];
      s0 += y0; q0 += y0 * y0; s1 += y1; q1 += y1 * y1;
    }
#pragma unroll
    for (int off = 16; off > 0; off >>= 1) {
      s0 += __shfl_xor(s0, off, 32);
      q0 += __shfl_xor(q0, off, 32);
      s1 += __shfl_xor(s1, off, 32);
      q1 += __shfl_xor(q1, off, 32);
    }
    const float inv_d = 1.0f / (float)DIM;
    float mu0 = s0 * inv_d, mu1 = s1 * inv_d;
    float i0  = rsqrtf(q0 * inv_d - mu0 * mu0 + 1e-5f);
    float i1  = rsqrtf(q1 * inv_d - mu1 * mu1 + 1e-5f);
    float* dst = out + (size_t)(vox0 + vx) * DIM;
#pragma unroll
    for (int i = 0; i < 12; ++i) {
      int d = lane + 32 * i;
      float y0 = Ol[r0 * OS + d], y1 = Ol[r1 * OS + d];
      dst[d] = 0.5f * ((y0 - mu0) * i0 + (y1 - mu1) * i1) * ln_w[d] + ln_b[d];
    }
  }
}

extern "C" void kernel_launch(void* const* d_in, const int* in_sizes, int n_in,
                              void* d_out, int out_size, void* d_ws, size_t ws_size,
                              hipStream_t stream) {
  const float* feat1 = (const float*)d_in[0];
  const float* feat2 = (const float*)d_in[1];
  const float* w_in  = (const float*)d_in[2];
  const float* b_in  = (const float*)d_in[3];
  const float* w_out = (const float*)d_in[4];
  const float* b_out = (const float*)d_in[5];
  const float* ln_w  = (const float*)d_in[6];
  const float* ln_b  = (const float*)d_in[7];
  float* outp = (float*)d_out;
  unsigned short* wbf = (unsigned short*)d_ws;   // [WIN_ELEMS | WOUT_ELEMS] bf16

  wcvt_kernel<<<1152, 256, 0, stream>>>(w_in, w_out, wbf);

  (void)hipFuncSetAttribute((const void*)attn_fused_kernel,
                            hipFuncAttributeMaxDynamicSharedMemorySize, LDS_TOTAL);
  const int M = (in_sizes[0] / DIM);             // B*N = 131072 voxels
  const int nWG = M / VOXPB;                     // 4096 workgroups
  attn_fused_kernel<<<nWG, 256, LDS_TOTAL, stream>>>(
      feat1, feat2, b_in, b_out, ln_w, ln_b, wbf, wbf + WIN_ELEMS, outp);
}
